// HybridSeqGConvSeq2One_75522704932946
// MI455X (gfx1250) — matrix operand order repaired and checked by reading, not compiled, not run on hardware
//
#include <hip/hip_runtime.h>
#include <hip/hip_bf16.h>
#include <stdint.h>

// ---------------- types / helpers ----------------
typedef __attribute__((ext_vector_type(16))) __bf16 bf16x16;
typedef __attribute__((ext_vector_type(8)))  float  f32x8;

#define HIDD 256

__device__ __forceinline__ unsigned short f2bf(float f) {
    unsigned u = __float_as_uint(f);
    u += 0x7fffu + ((u >> 16) & 1u);           // round-to-nearest-even
    return (unsigned short)(u >> 16);
}
__device__ __forceinline__ float bf2f(unsigned short b) {
    return __uint_as_float(((unsigned)b) << 16);
}
__device__ __forceinline__ float sigf(float x) { return 1.f / (1.f + expf(-x)); }

union FragU { bf16x16 v; uint4 u[2]; };

// ---------------- WMMA GEMM: C[M,N] = bf16(A[M,K]) @ packedB + bias ----------------
// A: bf16(u16) row-major MxK.  Bp: fragment-major packed bf16 (see pack_b_kernel).
// Each wave: one 16-row tile x 64 cols (4 wmma accumulators). 8 waves / block.
__global__ __launch_bounds__(256)
void gemm_bf16(const unsigned short* __restrict__ A,
               const unsigned short* __restrict__ Bp,
               const float* __restrict__ bias,
               float* __restrict__ C, int M, int N, int K)
{
    const int lane   = threadIdx.x & 31;
    const int wave   = threadIdx.x >> 5;
    const int mtile  = blockIdx.y * 8 + wave;          // uniform per wave
    if (mtile * 16 >= M) return;                       // whole-wave exit: EXEC stays all-1
    const int ntile0 = blockIdx.x * 4;
    const int half   = lane >> 4;
    const int l15    = lane & 15;
    const int m      = mtile * 16 + l15;
    const int ntiles = N >> 4;

    f32x8 acc0 = {0,0,0,0,0,0,0,0};
    f32x8 acc1 = {0,0,0,0,0,0,0,0};
    f32x8 acc2 = {0,0,0,0,0,0,0,0};
    f32x8 acc3 = {0,0,0,0,0,0,0,0};

    const unsigned short* arow = A + (size_t)m * K + half * 8;
    for (int k0 = 0; k0 < K; k0 += 32) {
        FragU af;
        af.u[0] = *(const uint4*)(arow);               // K = k0 + half*8 + 0..7
        af.u[1] = *(const uint4*)(arow + 16);          // K = k0 + 16 + half*8 + 0..7
        arow += 32;

        const unsigned short* bbase =
            Bp + (((size_t)(k0 >> 5) * ntiles + ntile0) * 32 + lane) * 16;
        if (k0 + 32 < K)                               // prefetch next k-tile of B
            __builtin_prefetch((const void*)(bbase + (size_t)ntiles * 512), 0, 0);

        FragU b0, b1, b2, b3;
        b0.u[0] = *(const uint4*)(bbase +    0); b0.u[1] = *(const uint4*)(bbase +    8);
        b1.u[0] = *(const uint4*)(bbase +  512); b1.u[1] = *(const uint4*)(bbase +  520);
        b2.u[0] = *(const uint4*)(bbase + 1024); b2.u[1] = *(const uint4*)(bbase + 1032);
        b3.u[0] = *(const uint4*)(bbase + 1536); b3.u[1] = *(const uint4*)(bbase + 1544);

        acc0 = __builtin_amdgcn_wmma_f32_16x16x32_bf16(false, af.v, false, b0.v, (short)0, acc0, false, false);
        acc1 = __builtin_amdgcn_wmma_f32_16x16x32_bf16(false, af.v, false, b1.v, (short)0, acc1, false, false);
        acc2 = __builtin_amdgcn_wmma_f32_16x16x32_bf16(false, af.v, false, b2.v, (short)0, acc2, false, false);
        acc3 = __builtin_amdgcn_wmma_f32_16x16x32_bf16(false, af.v, false, b3.v, (short)0, acc3, false, false);
    }

    float* crow = C + (size_t)(mtile * 16 + half * 8) * N;
    #pragma unroll
    for (int t = 0; t < 4; ++t) {
        f32x8 a = (t == 0) ? acc0 : (t == 1) ? acc1 : (t == 2) ? acc2 : acc3;
        const int n = (ntile0 + t) * 16 + l15;
        const float bv = bias ? bias[n] : 0.f;
        #pragma unroll
        for (int j = 0; j < 8; ++j)                    // VGPR j -> row m = mtile*16 + half*8 + j
            crow[(size_t)j * N + n] = a[j] + bv;
    }
}

// ---------------- weight staging ----------------
// B_f32[k0+c, n0+r] = W[r, c]   (W is torch [out,in] row-major -> B = W^T block)
__global__ void scatter_wT(const float* __restrict__ W, float* __restrict__ B,
                           int NB, int n0, int k0, int nrows, int ncols)
{
    int idx = blockIdx.x * blockDim.x + threadIdx.x;
    if (idx >= nrows * ncols) return;
    int r = idx / ncols, c = idx % ncols;
    B[(size_t)(k0 + c) * NB + (n0 + r)] = W[idx];
}

// pack f32 B[K,N] -> fragment-major bf16:
// idx -> (kt, nt, lane, i):  k = kt*32 + (i>>3)*16 + (lane>>4)*8 + (i&7),  n = nt*16 + (lane&15)
__global__ void pack_b_kernel(const float* __restrict__ B, unsigned short* __restrict__ Bp,
                              int K, int N)
{
    int idx = blockIdx.x * blockDim.x + threadIdx.x;
    if (idx >= K * N) return;
    int i    = idx & 15;
    int lane = (idx >> 4) & 31;
    int nt   = (idx >> 9) % (N >> 4);
    int kt   = (idx >> 9) / (N >> 4);
    int k = kt * 32 + (i >> 3) * 16 + (lane >> 4) * 8 + (i & 7);
    int n = nt * 16 + (lane & 15);
    Bp[idx] = f2bf(B[(size_t)k * N + n]);
}

__global__ void vec_add(const float* a, const float* b, float* o, int n) {
    int i = blockIdx.x * blockDim.x + threadIdx.x;
    if (i < n) o[i] = a[i] + b[i];
}

// ---------------- utility kernels ----------------
__global__ void zero_f32_kernel(float* p, long n) {
    long i = (long)blockIdx.x * blockDim.x + threadIdx.x;
    if (i < n) p[i] = 0.f;
}
__global__ void zero_cols_u16(unsigned short* D, int stride, int ofs, int dim, int Nn) {
    int idx = blockIdx.x * blockDim.x + threadIdx.x;
    if (idx >= Nn * dim) return;
    int n = idx / dim, j = idx % dim;
    D[(size_t)n * stride + ofs + j] = 0;
}
__global__ void copy_cols_bf16(const unsigned short* S, int sstride, int sofs,
                               unsigned short* D, int dstride, int dofs, int dim, int Nn) {
    int idx = blockIdx.x * blockDim.x + threadIdx.x;
    if (idx >= Nn * dim) return;
    int n = idx / dim, j = idx % dim;
    D[(size_t)n * dstride + dofs + j] = S[(size_t)n * sstride + sofs + j];
}
__global__ void cvt_cols_f32_bf16(const float* S, unsigned short* D,
                                  int dstride, int dofs, int dim, int Nn) {
    int idx = blockIdx.x * blockDim.x + threadIdx.x;
    if (idx >= Nn * dim) return;
    int n = idx / dim, j = idx % dim;
    D[(size_t)n * dstride + dofs + j] = f2bf(S[idx]);
}

// ---------------- embeddings ----------------
// stat: 16 -> 64 tanh -> 32 tanh ; writes fp32 copy + bf16 into XcatL cols 32..63
__global__ void stat_emb_kernel(const float* __restrict__ stat,
                                const float* __restrict__ w1, const float* __restrict__ b1,
                                const float* __restrict__ w2, const float* __restrict__ b2,
                                float* __restrict__ statf, unsigned short* __restrict__ XcatL, int Nn)
{
    int n = blockIdx.x * blockDim.x + threadIdx.x;
    if (n >= Nn) return;
    float x[16];
    #pragma unroll
    for (int k = 0; k < 16; ++k) x[k] = stat[n * 16 + k];
    float h[64];
    for (int j = 0; j < 64; ++j) {
        float a = b1[j];
        #pragma unroll
        for (int k = 0; k < 16; ++k) a += w1[j * 16 + k] * x[k];
        h[j] = tanhf(a);
    }
    for (int j = 0; j < 32; ++j) {
        float a = b2[j];
        for (int k = 0; k < 64; ++k) a += w2[j * 64 + k] * h[k];
        a = tanhf(a);
        statf[n * 32 + j] = a;
        XcatL[(size_t)n * 320 + 32 + j] = f2bf(a);
    }
}

// dyn (one timestep): 8 -> 64 tanh -> 32 tanh ; writes bf16 into XcatL cols 0..31
__global__ void dyn_emb_kernel(const float* __restrict__ dynT,
                               const float* __restrict__ w1, const float* __restrict__ b1,
                               const float* __restrict__ w2, const float* __restrict__ b2,
                               unsigned short* __restrict__ XcatL, int Nn)
{
    int n = blockIdx.x * blockDim.x + threadIdx.x;
    if (n >= Nn) return;
    float x[8];
    #pragma unroll
    for (int k = 0; k < 8; ++k) x[k] = dynT[n * 8 + k];
    float h[64];
    for (int j = 0; j < 64; ++j) {
        float a = b1[j];
        #pragma unroll
        for (int k = 0; k < 8; ++k) a += w1[j * 8 + k] * x[k];
        h[j] = tanhf(a);
    }
    for (int j = 0; j < 32; ++j) {
        float a = b2[j];
        for (int k = 0; k < 64; ++k) a += w2[j * 64 + k] * h[k];
        XcatL[(size_t)n * 320 + j] = f2bf(tanhf(a));
    }
}

// ---------------- recurrent cells ----------------
// LSTM step: Z[n,1024] (i|f|g|o) -> updates c (fp32) and h (bf16 into XcatL cols 64..319)
__global__ void lstm_update(const float* __restrict__ Z, float* __restrict__ c,
                            unsigned short* __restrict__ XcatL, int Nn)
{
    int idx = blockIdx.x * blockDim.x + threadIdx.x;
    if (idx >= Nn * HIDD) return;
    int n = idx >> 8, j = idx & 255;
    const float* z = Z + (size_t)n * 1024;
    float i = sigf(z[j]), f = sigf(z[256 + j]), g = tanhf(z[512 + j]), o = sigf(z[768 + j]);
    float cv = f * c[idx] + i * g;
    c[idx] = cv;
    XcatL[(size_t)n * 320 + 64 + j] = f2bf(o * tanhf(cv));
}

// ChebConv-LSTM gate update. c_in==null -> c=0 path (dec2).
__global__ void dec_gate(const float* __restrict__ Z, const float* __restrict__ c_in,
                         unsigned short* __restrict__ hb, int hstride, int hofs,
                         float* __restrict__ hf, int Nn)
{
    int idx = blockIdx.x * blockDim.x + threadIdx.x;
    if (idx >= Nn * HIDD) return;
    int n = idx >> 8, j = idx & 255;
    const float* z = Z + (size_t)n * 1024;
    float i = sigf(z[j]), f = sigf(z[256 + j]), g = tanhf(z[512 + j]), o = sigf(z[768 + j]);
    float cv = i * g;
    if (c_in) cv += f * c_in[idx];
    float h = o * tanhf(cv);
    if (hb) hb[(size_t)n * hstride + hofs + j] = f2bf(h);
    if (hf) hf[idx] = h;
}

// ---------------- graph ops ----------------
__global__ void deg_kernel(const int* __restrict__ src, const float* __restrict__ w,
                           float* __restrict__ deg, int nE)
{
    int e = blockIdx.x * blockDim.x + threadIdx.x;
    if (e >= nE) return;
    atomicAdd(&deg[src[e]], w[e]);
}
__global__ void norm_kernel(const int* __restrict__ src, const int* __restrict__ dst,
                            const float* __restrict__ w, const float* __restrict__ deg,
                            float* __restrict__ nrm, int nE)
{
    int e = blockIdx.x * blockDim.x + threadIdx.x;
    if (e >= nE) return;
    float ds = deg[src[e]], dd = deg[dst[e]];
    float a = ds > 0.f ? rsqrtf(ds) : 0.f;
    float b = dd > 0.f ? rsqrtf(dd) : 0.f;
    nrm[e] = -a * w[e] * b;
}
// out[dst,:] += nrm_e * X[src,:]  (X is bf16 node matrix with stride/col-offset)
__global__ void propagate(const int* __restrict__ src, const int* __restrict__ dst,
                          const float* __restrict__ nrm,
                          const unsigned short* __restrict__ X, int xstride, int xofs,
                          int dim, float* __restrict__ out, int nE)
{
    int idx = blockIdx.x * blockDim.x + threadIdx.x;
    int chunks = dim >> 2;
    int e = idx / chunks;
    if (e >= nE) return;
    int c0 = (idx % chunks) << 2;
    float w = nrm[e];
    int s = src[e], d = dst[e];
    const unsigned short* xs = X + (size_t)s * xstride + xofs + c0;
    float* op = out + (size_t)d * dim + c0;
    #pragma unroll
    for (int u = 0; u < 4; ++u) atomicAdd(op + u, w * bf2f(xs[u]));
}

// ---------------- heads ----------------
__global__ __launch_bounds__(128)
void heads_kernel(const float* __restrict__ h2, const float* __restrict__ statf,
                  const float* __restrict__ dw1, const float* __restrict__ db1,
                  const float* __restrict__ dw2, const float* __restrict__ db2,
                  const float* __restrict__ sw1, const float* __restrict__ sb1,
                  const float* __restrict__ sw2, const float* __restrict__ sb2,
                  float* __restrict__ out, int Nn)
{
    __shared__ float ds[288];
    __shared__ float red[128];
    const int n = blockIdx.x;
    const int t = threadIdx.x;
    for (int k = t; k < 256; k += 128) ds[k] = h2[(size_t)n * 256 + k];
    if (t < 32) ds[256 + t] = statf[n * 32 + t];
    __syncthreads();
    // discharge: relu(dstate @ dw1^T + db1) @ dw2^T + db2
    float a = db1[t];
    for (int k = 0; k < 288; ++k) a += ds[k] * dw1[t * 288 + k];
    red[t] = fmaxf(a, 0.f) * dw2[t];
    __syncthreads();
    for (int s = 64; s > 0; s >>= 1) { if (t < s) red[t] += red[t + s]; __syncthreads(); }
    if (t == 0) out[n] = red[0] + db2[0];
    __syncthreads();
    // swi: sigmoid(tanh(dstate @ sw1^T + sb1) @ sw2^T + sb2)
    if (t < 64) {
        float b = sb1[t];
        for (int k = 0; k < 288; ++k) b += ds[k] * sw1[t * 288 + k];
        red[t] = tanhf(b) * sw2[t];
    }
    __syncthreads();
    for (int s = 32; s > 0; s >>= 1) { if (t < s) red[t] += red[t + s]; __syncthreads(); }
    if (t == 0) out[Nn + n] = 1.f / (1.f + expf(-(red[0] + sb2[0])));
}

// ---------------- host side ----------------
extern "C" void kernel_launch(void* const* d_in, const int* in_sizes, int n_in,
                              void* d_out, int out_size, void* d_ws, size_t ws_size,
                              hipStream_t stream)
{
    (void)n_in; (void)out_size; (void)ws_size;
    const float* dynSeq = (const float*)d_in[0];
    const float* statF  = (const float*)d_in[1];
    const int*   eidx   = (const int*)d_in[2];
    const float* ew     = (const float*)d_in[3];
    const int Nn = in_sizes[1] / 16;
    const int S  = in_sizes[0] / (Nn * 8);
    const int E  = in_sizes[3];
    const int* src = eidx;
    const int* dst = eidx + E;

    // workspace layout
    char* w = (char*)d_ws;
    auto alloc = [&](size_t bytes) { void* p = (void*)w; w += (bytes + 255) & ~(size_t)255; return p; };
    unsigned short* XcatL = (unsigned short*)alloc((size_t)Nn * 320 * 2);
    float*          Z     = (float*)alloc((size_t)Nn * 1024 * 4);
    float*          cbuf  = (float*)alloc((size_t)Nn * 256 * 4);
    float*          statE = (float*)alloc((size_t)Nn * 32 * 4);
    unsigned short* Xcat1 = (unsigned short*)alloc((size_t)Nn * 640 * 2);
    unsigned short* Xcat2 = (unsigned short*)alloc((size_t)Nn * 512 * 2);
    float*          pxf   = (float*)alloc((size_t)Nn * 64 * 4);
    float*          phf   = (float*)alloc((size_t)Nn * 256 * 4);
    float*          h2f   = (float*)alloc((size_t)Nn * 256 * 4);
    float*          deg   = (float*)alloc((size_t)Nn * 4);
    float*          nrm   = (float*)alloc((size_t)E * 4);
    float*          Bstg  = (float*)alloc((size_t)640 * 1024 * 4);
    unsigned short* BpL   = (unsigned short*)alloc((size_t)320 * 1024 * 2);
    unsigned short* Bp1   = (unsigned short*)alloc((size_t)640 * 1024 * 2);
    unsigned short* Bp2   = (unsigned short*)alloc((size_t)512 * 1024 * 2);
    float*          biasL = (float*)alloc(1024 * 4);
    float*          bias1 = (float*)alloc(1024 * 4);
    float*          bias2 = (float*)alloc(1024 * 4);

    auto cdiv = [](long a, long b) { return (int)((a + b - 1) / b); };
    auto gemm = [&](const unsigned short* A, const unsigned short* Bp, const float* bias,
                    float* C, int M, int N, int K) {
        dim3 g(N / 64, cdiv(M / 16, 8));
        gemm_bf16<<<g, 256, 0, stream>>>(A, Bp, bias, C, M, N, K);
    };

    // ---- pack LSTM weights: B[320,1024] = [Wih^T ; Whh^T], bias = bih + bhh ----
    scatter_wT<<<cdiv(1024 * 64, 256), 256, 0, stream>>>((const float*)d_in[12], Bstg, 1024, 0, 0, 1024, 64);
    scatter_wT<<<cdiv(1024 * 256, 256), 256, 0, stream>>>((const float*)d_in[14], Bstg, 1024, 0, 64, 1024, 256);
    pack_b_kernel<<<cdiv(320 * 1024, 256), 256, 0, stream>>>(Bstg, BpL, 320, 1024);
    vec_add<<<cdiv(1024, 256), 256, 0, stream>>>((const float*)d_in[13], (const float*)d_in[15], biasL, 1024);

    // ---- graph normalization ----
    zero_f32_kernel<<<cdiv(Nn, 256), 256, 0, stream>>>(deg, Nn);
    deg_kernel<<<cdiv(E, 256), 256, 0, stream>>>(src, ew, deg, E);
    norm_kernel<<<cdiv(E, 256), 256, 0, stream>>>(src, dst, ew, deg, nrm, E);

    // ---- init state: c=0, h(bf16 cols 64..319 of XcatL)=0 ----
    zero_f32_kernel<<<cdiv((long)Nn * 256, 256), 256, 0, stream>>>(cbuf, (long)Nn * 256);
    zero_cols_u16<<<cdiv((long)Nn * 256, 256), 256, 0, stream>>>(XcatL, 320, 64, 256, Nn);

    // ---- static embedding (once) ----
    stat_emb_kernel<<<cdiv(Nn, 128), 128, 0, stream>>>(statF,
        (const float*)d_in[8], (const float*)d_in[9], (const float*)d_in[10], (const float*)d_in[11],
        statE, XcatL, Nn);

    // ---- encoder LSTM over time ----
    for (int t = 0; t < S; ++t) {
        dyn_emb_kernel<<<cdiv(Nn, 128), 128, 0, stream>>>(dynSeq + (size_t)t * Nn * 8,
            (const float*)d_in[4], (const float*)d_in[5], (const float*)d_in[6], (const float*)d_in[7],
            XcatL, Nn);
        gemm(XcatL, BpL, biasL, Z, Nn, 1024, 320);
        lstm_update<<<cdiv((long)Nn * 256, 256), 256, 0, stream>>>(Z, cbuf, XcatL, Nn);
    }

    // ---- pack dec1 weights: B[640,1024], rows [x|px|h|ph], gate order i,f,g,o ----
    for (int gi = 0; gi < 4; ++gi) {
        const float* xW = (const float*)d_in[16 + gi * 4];
        const float* xb = (const float*)d_in[17 + gi * 4];
        const float* hW = (const float*)d_in[18 + gi * 4];
        const float* hb = (const float*)d_in[19 + gi * 4];
        scatter_wT<<<cdiv(256 * 64, 256), 256, 0, stream>>>(xW,             Bstg, 1024, gi * 256, 0,   256, 64);
        scatter_wT<<<cdiv(256 * 64, 256), 256, 0, stream>>>(xW + 256 * 64,  Bstg, 1024, gi * 256, 64,  256, 64);
        scatter_wT<<<cdiv(256 * 256, 256), 256, 0, stream>>>(hW,            Bstg, 1024, gi * 256, 128, 256, 256);
        scatter_wT<<<cdiv(256 * 256, 256), 256, 0, stream>>>(hW + 256 * 256,Bstg, 1024, gi * 256, 384, 256, 256);
        vec_add<<<1, 256, 0, stream>>>(xb, hb, bias1 + gi * 256, 256);
    }
    pack_b_kernel<<<cdiv(640 * 1024, 256), 256, 0, stream>>>(Bstg, Bp1, 640, 1024);

    // ---- dec1 inputs: Xcat1 = [x_dec | L.x | h | L.h] (bf16) ----
    copy_cols_bf16<<<cdiv((long)Nn * 64, 256), 256, 0, stream>>>(XcatL, 320, 0,  Xcat1, 640, 0,   64,  Nn);
    copy_cols_bf16<<<cdiv((long)Nn * 256, 256), 256, 0, stream>>>(XcatL, 320, 64, Xcat1, 640, 128, 256, Nn);
    zero_f32_kernel<<<cdiv((long)Nn * 64, 256), 256, 0, stream>>>(pxf, (long)Nn * 64);
    propagate<<<cdiv((long)E * 16, 256), 256, 0, stream>>>(src, dst, nrm, XcatL, 320, 0, 64, pxf, E);
    cvt_cols_f32_bf16<<<cdiv((long)Nn * 64, 256), 256, 0, stream>>>(pxf, Xcat1, 640, 64, 64, Nn);
    zero_f32_kernel<<<cdiv((long)Nn * 256, 256), 256, 0, stream>>>(phf, (long)Nn * 256);
    propagate<<<cdiv((long)E * 64, 256), 256, 0, stream>>>(src, dst, nrm, XcatL, 320, 64, 256, phf, E);
    cvt_cols_f32_bf16<<<cdiv((long)Nn * 256, 256), 256, 0, stream>>>(phf, Xcat1, 640, 384, 256, Nn);

    gemm(Xcat1, Bp1, bias1, Z, Nn, 1024, 640);
    dec_gate<<<cdiv((long)Nn * 256, 256), 256, 0, stream>>>(Z, cbuf, Xcat2, 512, 0, (float*)nullptr, Nn);

    // ---- pack dec2 weights: B[512,1024], rows [h1|ph1] ----
    for (int gi = 0; gi < 4; ++gi) {
        const float* xW = (const float*)d_in[32 + gi * 4];
        const float* xb = (const float*)d_in[33 + gi * 4];
        const float* hb = (const float*)d_in[35 + gi * 4];
        scatter_wT<<<cdiv(256 * 256, 256), 256, 0, stream>>>(xW,             Bstg, 1024, gi * 256, 0,   256, 256);
        scatter_wT<<<cdiv(256 * 256, 256), 256, 0, stream>>>(xW + 256 * 256, Bstg, 1024, gi * 256, 256, 256, 256);
        vec_add<<<1, 256, 0, stream>>>(xb, hb, bias2 + gi * 256, 256);
    }
    pack_b_kernel<<<cdiv(512 * 1024, 256), 256, 0, stream>>>(Bstg, Bp2, 512, 1024);

    // ---- dec2: Xcat2 = [h1 | L.h1] ----
    zero_f32_kernel<<<cdiv((long)Nn * 256, 256), 256, 0, stream>>>(phf, (long)Nn * 256);
    propagate<<<cdiv((long)E * 64, 256), 256, 0, stream>>>(src, dst, nrm, Xcat2, 512, 0, 256, phf, E);
    cvt_cols_f32_bf16<<<cdiv((long)Nn * 256, 256), 256, 0, stream>>>(phf, Xcat2, 512, 256, 256, Nn);

    gemm(Xcat2, Bp2, bias2, Z, Nn, 1024, 512);
    dec_gate<<<cdiv((long)Nn * 256, 256), 256, 0, stream>>>(Z, (const float*)nullptr,
                                                            (unsigned short*)nullptr, 0, 0, h2f, Nn);

    // ---- heads ----
    heads_kernel<<<Nn, 128, 0, stream>>>(h2f, statE,
        (const float*)d_in[48], (const float*)d_in[49], (const float*)d_in[50], (const float*)d_in[51],
        (const float*)d_in[52], (const float*)d_in[53], (const float*)d_in[54], (const float*)d_in[55],
        (float*)d_out, Nn);
}
